// MAB_11759620456985
// MI455X (gfx1250) — matrix operand order repaired and checked by reading, not compiled, not run on hardware
//
#include <hip/hip_runtime.h>
#include <hip/hip_bf16.h>

// MAB (multihead attention block) fused pipeline for MI455X (gfx1250, wave32).
// All GEMM-shaped work goes through v_wmma_f32_16x16x32_f16 (f16 inputs, f32 acc).
// Inputs (Q, K, Wq/Wk/Wv/Wo) are converted f32->f16 ONCE up front so the GEMM
// hot loops are pure f16 b128 loads feeding WMMA (no per-reread conversion).
// Register blocking: 64x64 output per wave in GEMMs (16 WMMA : 16 b128 loads),
// 32 queries per wave in flash attention (each K/V fragment feeds 2 WMMAs; the
// online softmax uses one row per lane so EXEC stays all-ones for WMMA).
// Pipeline: cvt -> proj(q,k,v) -> flash attention (+q residual) -> LN0 -> FFN -> LN1

#define BATCH   4
#define NQ      1024
#define NK      4096
#define DIMV    1024
#define HEADS   8
#define DH      128

typedef __attribute__((ext_vector_type(16))) _Float16 v16h;
typedef __attribute__((ext_vector_type(4)))  _Float16 v4h;
typedef __attribute__((ext_vector_type(8)))  float    v8f;

// ---------------- WMMA fragment loaders (ISA 7.12.2 layouts, wave32) -------------
// A-matrix 16x32 f16: lane L holds row m=L&15; hi=L>>4.
//   VGPR v in 0..3 -> K = hi*8 + 2v ; VGPR v in 4..7 -> K = 16 + hi*8 + 2(v-4)
__device__ __forceinline__ v16h load_a_frag_f16(const _Float16* __restrict__ row) {
  const int hi8 = ((threadIdx.x & 31) >> 4) << 3;
  v16h f;
#pragma unroll
  for (int v = 0; v < 8; ++v) {
    const int k = ((v >> 2) << 4) + hi8 + ((v & 3) << 1);
    f[2 * v]     = row[k];
    f[2 * v + 1] = row[k + 1];
  }
  return f;
}
__device__ __forceinline__ v16h load_a_frag_f32(const float* __restrict__ row) {
  const int hi8 = ((threadIdx.x & 31) >> 4) << 3;
  v16h f;
#pragma unroll
  for (int v = 0; v < 8; ++v) {
    const int k = ((v >> 2) << 4) + hi8 + ((v & 3) << 1);
    f[2 * v]     = (_Float16)row[k];
    f[2 * v + 1] = (_Float16)row[k + 1];
  }
  return f;
}
// B-matrix 32x16 f16: lane L holds column n=L&15; K = (v>>2)*16 + (L>>4)*8 + 2(v&3)
__device__ __forceinline__ v16h load_b_frag_f16(const _Float16* __restrict__ row) {
  const int base = ((threadIdx.x & 31) >> 4) << 3;
  v16h f;
#pragma unroll
  for (int v = 0; v < 8; ++v) {
    const int k = ((v >> 2) << 4) + base + ((v & 3) << 1);
    f[2 * v]     = row[k];
    f[2 * v + 1] = row[k + 1];
  }
  return f;
}

#define WMMA_F16(A, B, C) \
  __builtin_amdgcn_wmma_f32_16x16x32_f16(false, (A), false, (B), (short)0, (C), false, false)

// ---------------- one-shot f32 -> f16 conversion (vectorized) --------------------
__global__ __launch_bounds__(256)
void cvt_kernel(const float* __restrict__ in, _Float16* __restrict__ out, int n4) {
  const int i = blockIdx.x * 256 + threadIdx.x;
  if (i < n4) {
    const float4 v = ((const float4*)in)[i];
    v4h o;
    o[0] = (_Float16)v.x; o[1] = (_Float16)v.y;
    o[2] = (_Float16)v.z; o[3] = (_Float16)v.w;
    ((v4h*)out)[i] = o;
  }
}

// ---------------- Projection: C = X @ W^T + b, 64x64 tile per wave ---------------
// Pure f16 operands (pre-converted), f32 accumulate + f32 bias.
// mode 0: q  -> outF32 row-major [M,1024] and outF16 row-major
// mode 1: k  -> outF16 row-major [M,1024]
// mode 2: v  -> outF16 transposed [B,H,DH,NK]  (row = (b*H+h)*DH+d, col = key)
__global__ __launch_bounds__(128)
void proj_kernel(const _Float16* __restrict__ X16, const _Float16* __restrict__ W16,
                 const float* __restrict__ bias, float* __restrict__ outF32,
                 _Float16* __restrict__ outF16, int mode) {
  const int lane = threadIdx.x & 31;
  const int wave = threadIdx.x >> 5;
  const int t    = blockIdx.x * 4 + wave;
  const int mt   = t >> 4;                 // 16 column stripes of 64 (N=1024)
  const int nt   = t & 15;
  const int row0 = mt << 6, col0 = nt << 6;
  const int n = lane & 15, hi8 = (lane >> 4) << 3;

  const _Float16* arow[4];
  const _Float16* brow[4];
#pragma unroll
  for (int i = 0; i < 4; ++i)
    arow[i] = X16 + (size_t)(row0 + i * 16 + n) * DIMV;
#pragma unroll
  for (int j = 0; j < 4; ++j)
    brow[j] = W16 + (size_t)(col0 + j * 16 + n) * DIMV;

  v8f acc[4][4] = {};
  for (int kk = 0; kk < DIMV; kk += 32) {
    v16h a[4], b[4];
#pragma unroll
    for (int i = 0; i < 4; ++i) a[i] = load_a_frag_f16(arow[i] + kk);
#pragma unroll
    for (int j = 0; j < 4; ++j) b[j] = load_b_frag_f16(brow[j] + kk);
#pragma unroll
    for (int i = 0; i < 4; ++i)
#pragma unroll
      for (int j = 0; j < 4; ++j) acc[i][j] = WMMA_F16(a[i], b[j], acc[i][j]);
  }

#pragma unroll
  for (int j = 0; j < 4; ++j) {
    const int col  = col0 + j * 16 + n;
    const float bv = bias[col];
#pragma unroll
    for (int i = 0; i < 4; ++i) {
#pragma unroll
      for (int r = 0; r < 8; ++r) {
        const int  mm = row0 + i * 16 + r + hi8;
        const float v = acc[i][j][r] + bv;
        if (mode == 0) {
          outF32[(size_t)mm * DIMV + col] = v;
          outF16[(size_t)mm * DIMV + col] = (_Float16)v;
        } else if (mode == 1) {
          outF16[(size_t)mm * DIMV + col] = (_Float16)v;
        } else {
          const int bb = mm >> 12, key = mm & (NK - 1);   // NK = 4096
          const int h = col >> 7, d = col & (DH - 1);
          outF16[(((size_t)(bb * HEADS + h) * DH + d) << 12) + key] = (_Float16)v;
        }
      }
    }
  }
}

// ---------------- Flash attention: oh = qh + softmax(q k^T / 32) v ----------------
// One wave per (b, h, 32-query tile). NK streamed in 32-key blocks.
__global__ __launch_bounds__(128)
void attn_kernel(const _Float16* __restrict__ qF16, const float* __restrict__ qF32,
                 const _Float16* __restrict__ kF16, const _Float16* __restrict__ vT16,
                 float* __restrict__ oh) {
  __shared__ float s_p[4][32 * 32];   // per-wave probability tile (32 q x 32 k)
  __shared__ float s_f[4][32];        // per-row rescale factor
  __shared__ float s_l[4][32];        // per-row denominator

  const int lane = threadIdx.x & 31;
  const int w    = threadIdx.x >> 5;
  const int t    = blockIdx.x * 4 + w;        // 0..1023
  const int qt   = t & 31;                    // NQ/32 tiles
  const int h    = (t >> 5) & 7;
  const int b    = t >> 8;
  const int q0   = qt << 5;
  const int n    = lane & 15, hi8 = (lane >> 4) << 3;

  // Preload Q A-fragments: 2 m-tiles x full dh=128 strip (4 chunks of K=32).
  v16h aq[2][4];
#pragma unroll
  for (int mt2 = 0; mt2 < 2; ++mt2) {
    const _Float16* qrow =
        qF16 + (size_t)(b * NQ + q0 + mt2 * 16 + n) * DIMV + h * DH;
#pragma unroll
    for (int c = 0; c < 4; ++c) aq[mt2][c] = load_a_frag_f16(qrow + c * 32);
  }

  v8f accO[2][8] = {};              // 32x128 output accumulator
  float m_i = -1e30f, l_i = 0.f;    // one query row per lane (row = lane)

  for (int kk = 0; kk < NK; kk += 32) {
    // Prefetch next key/value block (global_prefetch_b8).
    if (kk + 32 < NK) {
      __builtin_prefetch(
          (const void*)(kF16 + (size_t)(b * NK + kk + 32 + n) * DIMV + h * DH), 0, 3);
      __builtin_prefetch(
          (const void*)(vT16 + (((size_t)(b * HEADS + h) * DH + lane * 4) << 12) + kk + 32),
          0, 3);
    }
    // ---- scores S[32x32] = q . k^T : 16 WMMA, K fragments reused by 2 m-tiles ----
    v8f sacc[2][2] = {};
#pragma unroll
    for (int nt2 = 0; nt2 < 2; ++nt2) {
#pragma unroll
      for (int c = 0; c < 4; ++c) {
        const _Float16* kr =
            kF16 + (size_t)(b * NK + kk + nt2 * 16 + n) * DIMV + h * DH + c * 32;
        const v16h kb = load_b_frag_f16(kr);
        sacc[0][nt2] = WMMA_F16(aq[0][c], kb, sacc[0][nt2]);
        sacc[1][nt2] = WMMA_F16(aq[1][c], kb, sacc[1][nt2]);
      }
    }
    // scale 1/sqrt(1024), spill to LDS (C layout: row = r+8*hi, col = lane n)
#pragma unroll
    for (int mt2 = 0; mt2 < 2; ++mt2)
#pragma unroll
      for (int nt2 = 0; nt2 < 2; ++nt2)
#pragma unroll
        for (int r = 0; r < 8; ++r)
          s_p[w][(mt2 * 16 + r + hi8) * 32 + nt2 * 16 + n] =
              sacc[mt2][nt2][r] * 0.03125f;
    asm volatile("s_wait_dscnt 0" ::: "memory");

    // ---- online softmax: one row per lane, all 32 lanes active ----
    {
      float mx = m_i;
#pragma unroll
      for (int c = 0; c < 32; ++c) mx = fmaxf(mx, s_p[w][lane * 32 + c]);
      const float fsc = __expf(m_i - mx);
      float sum = 0.f;
#pragma unroll
      for (int c = 0; c < 32; ++c) {
        const float p = __expf(s_p[w][lane * 32 + c] - mx);
        s_p[w][lane * 32 + c] = p;
        sum += p;
      }
      l_i = l_i * fsc + sum;
      m_i = mx;
      s_f[w][lane] = fsc;
    }
    asm volatile("s_wait_dscnt 0" ::: "memory");

    // ---- rescale running output ----
#pragma unroll
    for (int mt2 = 0; mt2 < 2; ++mt2) {
      float fr[8];
#pragma unroll
      for (int r = 0; r < 8; ++r) fr[r] = s_f[w][mt2 * 16 + r + hi8];
#pragma unroll
      for (int d = 0; d < 8; ++d)
#pragma unroll
        for (int r = 0; r < 8; ++r) accO[mt2][d][r] *= fr[r];
    }

    // ---- O += P @ V : 16 WMMA, V fragments reused by 2 m-tiles ----
    const v16h pf0 = load_a_frag_f32(&s_p[w][n * 32]);
    const v16h pf1 = load_a_frag_f32(&s_p[w][(16 + n) * 32]);
#pragma unroll
    for (int d = 0; d < 8; ++d) {
      const _Float16* vr =
          vT16 + (((size_t)(b * HEADS + h) * DH + d * 16 + n) << 12) + kk;
      const v16h vb = load_b_frag_f16(vr);
      accO[0][d] = WMMA_F16(pf0, vb, accO[0][d]);
      accO[1][d] = WMMA_F16(pf1, vb, accO[1][d]);
    }
  }

  // ---- epilogue: oh = qF32 + accO / l ----
  s_l[w][lane] = l_i;
  asm volatile("s_wait_dscnt 0" ::: "memory");
#pragma unroll
  for (int mt2 = 0; mt2 < 2; ++mt2) {
    float linv[8];
#pragma unroll
    for (int r = 0; r < 8; ++r) linv[r] = 1.f / s_l[w][mt2 * 16 + r + hi8];
#pragma unroll
    for (int d = 0; d < 8; ++d) {
#pragma unroll
      for (int r = 0; r < 8; ++r) {
        const size_t row = (size_t)(b * NQ + q0 + mt2 * 16 + r + hi8);
        const int    col = h * DH + d * 16 + n;
        oh[row * DIMV + col] = qF32[row * DIMV + col] + accO[mt2][d][r] * linv[r];
      }
    }
  }
}

// ---------------- LayerNorm over 1024 cols; optional f16 copy --------------------
__global__ __launch_bounds__(256)
void ln_kernel(const float* __restrict__ in, const float* __restrict__ gam,
               const float* __restrict__ bet, float* __restrict__ out32,
               _Float16* __restrict__ out16) {
  const int row = blockIdx.x;
  const float* x = in + (size_t)row * DIMV;
  float vals[4], s = 0.f, ss = 0.f;
#pragma unroll
  for (int i = 0; i < 4; ++i) {
    const float v = x[threadIdx.x + i * 256];
    vals[i] = v; s += v; ss += v * v;
  }
#pragma unroll
  for (int off = 16; off > 0; off >>= 1) {
    s  += __shfl_xor(s, off);
    ss += __shfl_xor(ss, off);
  }
  __shared__ float rs[8], rss[8];
  if ((threadIdx.x & 31) == 0) { rs[threadIdx.x >> 5] = s; rss[threadIdx.x >> 5] = ss; }
  __syncthreads();
  float ts = 0.f, tss = 0.f;
#pragma unroll
  for (int wv = 0; wv < 8; ++wv) { ts += rs[wv]; tss += rss[wv]; }
  const float mu  = ts * (1.f / DIMV);
  const float var = tss * (1.f / DIMV) - mu * mu;
  const float inv = rsqrtf(var + 1e-5f);
#pragma unroll
  for (int i = 0; i < 4; ++i) {
    const int c = threadIdx.x + i * 256;
    const float y = (vals[i] - mu) * inv * gam[c] + bet[c];
    out32[(size_t)row * DIMV + c] = y;
    if (out16) out16[(size_t)row * DIMV + c] = (_Float16)y;
  }
}

// ---------------- FFN: z = x + relu(x @ Wo^T + bo), 64x64 tile per wave ----------
__global__ __launch_bounds__(128)
void ffn_kernel(const _Float16* __restrict__ x16, const float* __restrict__ x32,
                const _Float16* __restrict__ Wo16, const float* __restrict__ bo,
                float* __restrict__ z) {
  const int lane = threadIdx.x & 31;
  const int wave = threadIdx.x >> 5;
  const int t    = blockIdx.x * 4 + wave;
  const int mt   = t >> 4, nt = t & 15;
  const int row0 = mt << 6, col0 = nt << 6;
  const int n = lane & 15, hi8 = (lane >> 4) << 3;

  const _Float16* arow[4];
  const _Float16* brow[4];
#pragma unroll
  for (int i = 0; i < 4; ++i)
    arow[i] = x16 + (size_t)(row0 + i * 16 + n) * DIMV;
#pragma unroll
  for (int j = 0; j < 4; ++j)
    brow[j] = Wo16 + (size_t)(col0 + j * 16 + n) * DIMV;

  v8f acc[4][4] = {};
  for (int kk = 0; kk < DIMV; kk += 32) {
    v16h a[4], b[4];
#pragma unroll
    for (int i = 0; i < 4; ++i) a[i] = load_a_frag_f16(arow[i] + kk);
#pragma unroll
    for (int j = 0; j < 4; ++j) b[j] = load_b_frag_f16(brow[j] + kk);
#pragma unroll
    for (int i = 0; i < 4; ++i)
#pragma unroll
      for (int j = 0; j < 4; ++j) acc[i][j] = WMMA_F16(a[i], b[j], acc[i][j]);
  }

#pragma unroll
  for (int j = 0; j < 4; ++j) {
    const int col  = col0 + j * 16 + n;
    const float bv = bo[col];
#pragma unroll
    for (int i = 0; i < 4; ++i) {
#pragma unroll
      for (int r = 0; r < 8; ++r) {
        const size_t mm = (size_t)(row0 + i * 16 + r + hi8);
        const float  y  = fmaxf(acc[i][j][r] + bv, 0.f);
        z[mm * DIMV + col] = x32[mm * DIMV + col] + y;
      }
    }
  }
}

// ---------------- host-side launch ----------------------------------------------
extern "C" void kernel_launch(void* const* d_in, const int* in_sizes, int n_in,
                              void* d_out, int out_size, void* d_ws, size_t ws_size,
                              hipStream_t stream) {
  const float* Q  = (const float*)d_in[0];
  const float* K  = (const float*)d_in[1];
  const float* Wq = (const float*)d_in[2];
  const float* bq = (const float*)d_in[3];
  const float* Wk = (const float*)d_in[4];
  const float* bk = (const float*)d_in[5];
  const float* Wv = (const float*)d_in[6];
  const float* bv = (const float*)d_in[7];
  const float* Wo = (const float*)d_in[8];
  const float* bo = (const float*)d_in[9];
  const float* g0 = (const float*)d_in[10];
  const float* b0 = (const float*)d_in[11];
  const float* g1 = (const float*)d_in[12];
  const float* b1 = (const float*)d_in[13];
  float* out = (float*)d_out;

  char* ws = (char*)d_ws;
  const size_t SZ_KV16 = (size_t)BATCH * NK * DIMV * 2;  // 33.55 MB
  const size_t SZ_Q32  = (size_t)BATCH * NQ * DIMV * 4;  // 16.78 MB
  const size_t SZ_Q16  = (size_t)BATCH * NQ * DIMV * 2;  //  8.39 MB
  const size_t SZ_W16  = (size_t)DIMV * DIMV * 2;        //  2.10 MB
  // persistent through attention:
  _Float16* kF16 = (_Float16*)(ws);                                  // proj-k out
  _Float16* vT16 = (_Float16*)(ws + SZ_KV16);                        // proj-v out
  float*    qF32 = (float*)   (ws + 2 * SZ_KV16);                    // proj-q out
  _Float16* qF16 = (_Float16*)(ws + 2 * SZ_KV16 + SZ_Q32);
  char*     ohB  =             ws + 2 * SZ_KV16 + SZ_Q32 + SZ_Q16;   // attn out
  float*    oh   = (float*)ohB;
  // transient f16 input copies: Qin/Wq/Wk/Wv overlay the (not yet written) oh
  // region; Kin/Wo sit after it (Wo16 survives until the FFN).
  _Float16* Qin16 = (_Float16*)(ohB);
  _Float16* Wq16  = (_Float16*)(ohB + SZ_Q16);
  _Float16* Wk16  = (_Float16*)(ohB + SZ_Q16 + SZ_W16);
  _Float16* Wv16  = (_Float16*)(ohB + SZ_Q16 + 2 * SZ_W16);
  _Float16* Kin16 = (_Float16*)(ws + 2 * SZ_KV16 + 2 * SZ_Q32 + SZ_Q16);
  _Float16* Wo16  = (_Float16*)(ws + 3 * SZ_KV16 + 2 * SZ_Q32 + SZ_Q16);
  // reuse once producers are consumed:
  float*    x032 = (float*)   (ws);                  // over kF16 (dead after attn)
  _Float16* x016 = (_Float16*)(ws + SZ_Q32);         // still inside old kF16 region
  float*    z    = (float*)   (ws + SZ_KV16);        // over vT16 (dead after attn)

  dim3 blk(128);
  // one-shot f32 -> f16 conversions of all GEMM operands
  const int n4Q = BATCH * NQ * DIMV / 4;   // 1,048,576
  const int n4K = BATCH * NK * DIMV / 4;   // 4,194,304
  const int n4W = DIMV * DIMV / 4;         //   262,144
  cvt_kernel<<<n4Q / 256, 256, 0, stream>>>(Q,  Qin16, n4Q);
  cvt_kernel<<<n4K / 256, 256, 0, stream>>>(K,  Kin16, n4K);
  cvt_kernel<<<n4W / 256, 256, 0, stream>>>(Wq, Wq16,  n4W);
  cvt_kernel<<<n4W / 256, 256, 0, stream>>>(Wk, Wk16,  n4W);
  cvt_kernel<<<n4W / 256, 256, 0, stream>>>(Wv, Wv16,  n4W);
  cvt_kernel<<<n4W / 256, 256, 0, stream>>>(Wo, Wo16,  n4W);
  // q/k/v projections: 64x64 tile per wave, pure f16 operands
  proj_kernel<<<256,  blk, 0, stream>>>(Qin16, Wq16, bq, qF32,   qF16, 0);  // 1024 tiles
  proj_kernel<<<1024, blk, 0, stream>>>(Kin16, Wk16, bk, nullptr, kF16, 1); // 4096 tiles
  proj_kernel<<<1024, blk, 0, stream>>>(Kin16, Wv16, bv, nullptr, vT16, 2);
  // flash attention + residual: 32 queries per wave -> 1024 waves
  attn_kernel<<<256, blk, 0, stream>>>(qF16, qF32, kF16, vT16, oh);
  // LN0 (emit f16 copy for FFN GEMM)
  ln_kernel<<<BATCH * NQ, 256, 0, stream>>>(oh, g0, b0, x032, x016);
  // FFN with relu + residual
  ffn_kernel<<<256, blk, 0, stream>>>(x016, x032, Wo16, bo, z);
  // LN1 -> output
  ln_kernel<<<BATCH * NQ, 256, 0, stream>>>(z, g1, b1, out, nullptr);
}
